// NlsConv2d_31224412242789
// MI455X (gfx1250) — matrix operand order repaired and checked by reading, not compiled, not run on hardware
//
#include <hip/hip_runtime.h>

// ---------------------------------------------------------------------------
// Non-local search + weighted-stack conv3d for MI455X (gfx1250, wave32, WMMA)
//
// Shapes: B=4, C=32, H=W=256, WS=9 (pad 4), K_NEIGH=10, Cq=Cout=32.
//
// Pipeline (all scratch L2-resident, ~61 MB << 192 MB L2):
//   0a) xcast   : x (f32, NCHW) -> xh (f16, NHWC)  [channel = WMMA K dim]
//   0b) pack    : wq/wk/conv_w -> f16 A-fragments in exact WMMA lane layout
//   1)  proj_qk : q/k = 1x1 conv, WMMA f16, packed f16 NHWC outputs
//   2)  nls_topk: 81 window scores -> stable top-10 -> softmax(v/10)
//   3)  nls_conv: implicit GEMM (Kred = 32c x 10k x 9tap = 2880);
//                 A-fragments staged in LDS per block, 32-pixel N-blocking:
//                 each A fragment feeds 2 WMMAs.
// ---------------------------------------------------------------------------

typedef __attribute__((ext_vector_type(16))) _Float16 v16h;
typedef __attribute__((ext_vector_type(8)))  _Float16 v8h;
typedef __attribute__((ext_vector_type(8)))  float    v8f;

#define NB   4
#define NC   32
#define NH   256
#define NW   256
#define HW   (NH * NW)          // 65536
#define NPIX (NB * HW)          // 262144
#define WS   9
#define PAD  4
#define KN   10

// workspace byte offsets
#define WS_XH    ((size_t)0)          // 16 MiB  f16 NHWC x
#define WS_Q     ((size_t)16777216)   // 16 MiB  f16 NHWC q
#define WS_K     ((size_t)33554432)   // 16 MiB  f16 NHWC k
#define WS_TOPW  ((size_t)50331648)   // 10 MiB  f32 weights [pix][10]
#define WS_TOPI  ((size_t)60817408)   // 2.5 MiB u8 indices  [pix][10]
#define WS_AW    ((size_t)63438848)   // 180 KiB f16 conv A-frags [10][9][2][32][16]
#define WS_AQ    ((size_t)63623168)   // 4 KiB   f16 q-proj A-frags [2][32][16]
#define WS_AK    ((size_t)63627264)   // 4 KiB   f16 k-proj A-frags [2][32][16]

__device__ __forceinline__ int clampi(int v, int lo, int hi) {
    return v < lo ? lo : (v > hi ? hi : v);
}

// ---------------------------------------------------------------------------
// Prep 0a: x (f32 NCHW) -> xh (f16 NHWC). One thread per pixel.
// ---------------------------------------------------------------------------
__global__ void xcast_kernel(const float* __restrict__ x, _Float16* __restrict__ xh)
{
    const int p  = blockIdx.x * blockDim.x + threadIdx.x;   // 0..NPIX-1
    const int b  = p >> 16;
    const int hw = p & (HW - 1);
    const float* src = x + (size_t)b * NC * HW + hw;
    _Float16*    dst = xh + (size_t)p * NC;
    #pragma unroll
    for (int c = 0; c < NC; ++c) dst[c] = (_Float16)src[(size_t)c * HW];
}

// ---------------------------------------------------------------------------
// Prep 0b: pack weights into per-lane WMMA A fragments (f16, 16 halfs/lane).
// A layout (16x32 f16): lane m = lane%16; lo-lanes K 0..15,
// hi-lanes K 16..31.  conv frags: [(kk*9+tap) 0..89][mt 0..1][lane][16].
// ---------------------------------------------------------------------------
__global__ void pack_kernel(const float* __restrict__ wq, const float* __restrict__ wk,
                            const float* __restrict__ cw,
                            _Float16* __restrict__ aw,
                            _Float16* __restrict__ aq, _Float16* __restrict__ ak)
{
    const int t = blockIdx.x * blockDim.x + threadIdx.x;
    if (t >= 5888) return;
    if (t < 5760) {                       // conv_w fragments: 90*2*32
        const int lane = t & 31;
        const int mt   = (t >> 5) & 1;
        const int wofs = t >> 6;          // kk*9 + dy*3 + dx
        const int m     = lane & 15;
        const int kbase = (lane >= 16) ? 16 : 0;
        _Float16* dst = aw + (size_t)t * 16;
        const float* src = cw + (size_t)((mt * 16 + m) * NC) * 90 + wofs;
        #pragma unroll
        for (int i = 0; i < 8; ++i) {
            dst[i]     = (_Float16)src[(size_t)(kbase + i)      * 90];
            dst[8 + i] = (_Float16)src[(size_t)(8 + kbase + i)  * 90];
        }
    } else {                              // wq / wk fragments: 2 * (2*32)
        const int u    = t - 5760;        // 0..127
        const int isK  = u >> 6;          // 0 -> q, 1 -> k
        const int v    = u & 63;
        const int lane = v & 31;
        const int mt   = v >> 5;
        const int m     = lane & 15;
        const int kbase = (lane >= 16) ? 16 : 0;
        _Float16* dst = (isK ? ak : aq) + (size_t)v * 16;
        const float* src = (isK ? wk : wq) + (mt * 16 + m) * NC;
        #pragma unroll
        for (int i = 0; i < 8; ++i) {
            dst[i]     = (_Float16)src[kbase + i];
            dst[8 + i] = (_Float16)src[8 + kbase + i];
        }
    }
}

// ---------------------------------------------------------------------------
// Kernel 1: q/k projections.  One wave = one 16-pixel tile, 4 WMMAs.
// ---------------------------------------------------------------------------
__global__ void proj_qk_kernel(const _Float16* __restrict__ xh,
                               const _Float16* __restrict__ aqf,
                               const _Float16* __restrict__ akf,
                               const float* __restrict__ bq, const float* __restrict__ bk,
                               _Float16* __restrict__ qbuf, _Float16* __restrict__ kbuf)
{
    const int wave = threadIdx.x >> 5;
    const int lane = threadIdx.x & 31;
    const int tile = blockIdx.x * (blockDim.x >> 5) + wave;

    const int n = lane & 15;
    const int p = tile * 16 + n;
    const bool hi   = lane >= 16;
    const int khalf = hi ? 16 : 0;
    const int rbase = hi ? 8 : 0;

    // B tile: 16 contiguous f16 channels of pixel p (one 32B load)
    const v16h bmat = *(const v16h*)(xh + (size_t)p * NC + khalf);

    const v16h* aqv = (const v16h*)aqf;
    const v16h* akv = (const v16h*)akf;

    #pragma unroll
    for (int mt = 0; mt < 2; ++mt) {
        const v16h aq = aqv[mt * 32 + lane];
        const v16h ak = akv[mt * 32 + lane];
        v8f cq = {}, ck = {};
        cq = __builtin_amdgcn_wmma_f32_16x16x32_f16(false, aq, false, bmat,
                                                    (short)0, cq, false, false);
        ck = __builtin_amdgcn_wmma_f32_16x16x32_f16(false, ak, false, bmat,
                                                    (short)0, ck, false, false);
        // 8 D-rows per lane = 8 consecutive channels of pixel p -> packed store
        const int oc0 = mt * 16 + rbase;
        v8h oq, ok;
        #pragma unroll
        for (int i = 0; i < 8; ++i) {
            oq[i] = (_Float16)(cq[i] + bq[oc0 + i]);
            ok[i] = (_Float16)(ck[i] + bk[oc0 + i]);
        }
        *(v8h*)(qbuf + (size_t)p * NC + oc0) = oq;
        *(v8h*)(kbuf + (size_t)p * NC + oc0) = ok;
    }
}

// ---------------------------------------------------------------------------
// Kernel 2: 81 window dot-products, stable top-10, softmax(v/10).
// Channel-last f16 -> fully contiguous 64B reads per candidate.
// ---------------------------------------------------------------------------
__global__ void nls_topk_kernel(const _Float16* __restrict__ qbuf,
                                const _Float16* __restrict__ kbuf,
                                float* __restrict__ topw,
                                unsigned char* __restrict__ topi)
{
    const int p  = blockIdx.x * blockDim.x + threadIdx.x;
    const int b  = p >> 16;
    const int hw = p & (HW - 1);
    const int h  = hw >> 8;
    const int w  = hw & (NW - 1);

    float q[NC];
    {
        const v16h q0 = *(const v16h*)(qbuf + (size_t)p * NC);
        const v16h q1 = *(const v16h*)(qbuf + (size_t)p * NC + 16);
        #pragma unroll
        for (int i = 0; i < 16; ++i) { q[i] = (float)q0[i]; q[16 + i] = (float)q1[i]; }
    }

    float tv[KN];
    int   ti[KN];
    #pragma unroll
    for (int j = 0; j < KN; ++j) { tv[j] = -3.0e38f; ti[j] = 0; }

    for (int o = 0; o < WS * WS; ++o) {
        const int ky = clampi(h + o / WS - PAD, 0, NH - 1);
        const int kx = clampi(w + o % WS - PAD, 0, NW - 1);
        const size_t kp = (size_t)((b << 16) + (ky << 8) + kx) * NC;
        const v16h k0 = *(const v16h*)(kbuf + kp);
        const v16h k1 = *(const v16h*)(kbuf + kp + 16);
        float s = 0.0f;
        #pragma unroll
        for (int c = 0; c < 16; ++c) s += q[c] * (float)k0[c];
        #pragma unroll
        for (int c = 0; c < 16; ++c) s += q[16 + c] * (float)k1[c];
        // stable descending insertion (strict > keeps earliest index on ties)
        int pos = KN;
        while (pos > 0 && s > tv[pos - 1]) --pos;
        if (pos < KN) {
            for (int j = KN - 1; j > pos; --j) { tv[j] = tv[j - 1]; ti[j] = ti[j - 1]; }
            tv[pos] = s; ti[pos] = o;
        }
    }

    float e[KN], sum = 0.0f;
    #pragma unroll
    for (int j = 0; j < KN; ++j) { e[j] = __expf((tv[j] - tv[0]) * 0.1f); sum += e[j]; }
    const float inv = 1.0f / sum;
    float*         tw = topw + (size_t)p * KN;
    unsigned char* tx = topi + (size_t)p * KN;
    #pragma unroll
    for (int j = 0; j < KN; ++j) { tw[j] = e[j] * inv; tx[j] = (unsigned char)ti[j]; }
}

// ---------------------------------------------------------------------------
// Kernel 3: fused weighted-stack + conv3d implicit GEMM.
// Block = 128 threads (4 waves); each wave owns a 32-pixel tile (2 N-subtiles).
// Per kk step: block stages the 9-tap A-fragment slice (18 KB) into LDS once,
// then every wave runs 9 taps x (2 Mtiles x 2 Ntiles) = 36 accumulating WMMAs,
// reading A via ds_load_b128. B columns are gathered, weighted x (f16 NHWC).
// ---------------------------------------------------------------------------
#define KSLICE_HALFS (9 * 2 * 32 * 16)      // 9216 halfs = 18 KiB per kk

__global__ void nls_conv_kernel(const _Float16* __restrict__ xh,
                                const _Float16* __restrict__ awf,  // [10][9][2][32][16]
                                const float* __restrict__ cb,
                                const float* __restrict__ topw,
                                const unsigned char* __restrict__ topi,
                                float* __restrict__ out)
{
    __shared__ __align__(32) _Float16 alds[KSLICE_HALFS];

    const int tid  = threadIdx.x;           // 0..127
    const int wave = tid >> 5;
    const int lane = tid & 31;
    const int tile = blockIdx.x * 4 + wave; // 32-pixel tile id

    const int p0  = tile * 32;
    const int b   = p0 >> 16;
    const int hw0 = p0 & (HW - 1);
    const int h   = hw0 >> 8;
    const int w0  = hw0 & (NW - 1);
    const bool hi   = lane >= 16;
    const int khalf = hi ? 16 : 0;
    const int n     = lane & 15;

    v8f acc0a = {}, acc1a = {};             // N-subtile 0 (pixels w0 .. w0+15)
    v8f acc0b = {}, acc1b = {};             // N-subtile 1 (pixels w0+16 .. w0+31)

    for (int kk = 0; kk < KN; ++kk) {
        // ---- stage this kk's A fragments into LDS (coalesced, 16B chunks) ----
        __syncthreads();                     // previous iteration done reading
        {
            const v8h* src = (const v8h*)(awf + (size_t)kk * KSLICE_HALFS);
            v8h*       dst = (v8h*)alds;
            #pragma unroll
            for (int j = 0; j < 9; ++j)      // 1152 chunks / 128 threads
                dst[j * 128 + tid] = src[j * 128 + tid];
        }
        // prefetch next neighbor's slice while we compute (global_prefetch_b8)
        if (kk + 1 < KN)
            __builtin_prefetch(awf + (size_t)(kk + 1) * KSLICE_HALFS + tid * 64, 0, 1);
        __syncthreads();

        #pragma unroll
        for (int dy = 0; dy < 3; ++dy) {
            #pragma unroll
            for (int dx = 0; dx < 3; ++dx) {
                const int tap = dy * 3 + dx;
                const int y   = h + dy - 1;
                // ---- B columns: weighted gather at tap, zero-pad edges ----
                v16h bmatA = {}, bmatB = {};
                #pragma unroll
                for (int sub = 0; sub < 2; ++sub) {
                    const int xx = w0 + sub * 16 + n + dx - 1;
                    if (y >= 0 && y < NH && xx >= 0 && xx < NW) {
                        const int pi = (b << 16) + (y << 8) + xx;
                        const float wgt = topw[(size_t)pi * KN + kk];
                        const int   idx = (int)topi[(size_t)pi * KN + kk];
                        const int gy = clampi(y  + idx / WS - PAD, 0, NH - 1);
                        const int gx = clampi(xx + idx % WS - PAD, 0, NW - 1);
                        const int pg = (b << 16) + (gy << 8) + gx;
                        const v16h xv = *(const v16h*)(xh + (size_t)pg * NC + khalf);
                        const _Float16 wh = (_Float16)wgt;
                        v16h wsplat;
                        #pragma unroll
                        for (int i = 0; i < 16; ++i) wsplat[i] = wh;
                        if (sub == 0) bmatA = xv * wsplat;  // 8x v_pk_mul_f16
                        else          bmatB = xv * wsplat;
                    }
                }
                // ---- A fragments from LDS (ds_load_b128), reused by 4 WMMAs
                const v16h a0 = *(const v16h*)(alds + ((tap * 2 + 0) * 32 + lane) * 16);
                const v16h a1 = *(const v16h*)(alds + ((tap * 2 + 1) * 32 + lane) * 16);
                acc0a = __builtin_amdgcn_wmma_f32_16x16x32_f16(false, a0, false, bmatA,
                                                               (short)0, acc0a, false, false);
                acc1a = __builtin_amdgcn_wmma_f32_16x16x32_f16(false, a1, false, bmatA,
                                                               (short)0, acc1a, false, false);
                acc0b = __builtin_amdgcn_wmma_f32_16x16x32_f16(false, a0, false, bmatB,
                                                               (short)0, acc0b, false, false);
                acc1b = __builtin_amdgcn_wmma_f32_16x16x32_f16(false, a1, false, bmatB,
                                                               (short)0, acc1b, false, false);
            }
        }
    }

    #pragma unroll
    for (int i = 0; i < 8; ++i) {
        const int r = (hi ? 8 : 0) + i;
        const size_t row0 = ((size_t)b * NC + r)      * HW + h * NW;
        const size_t row1 = ((size_t)b * NC + 16 + r) * HW + h * NW;
        out[row0 + (w0 + n)]      = acc0a[i] + cb[r];
        out[row1 + (w0 + n)]      = acc1a[i] + cb[16 + r];
        out[row0 + (w0 + 16 + n)] = acc0b[i] + cb[r];
        out[row1 + (w0 + 16 + n)] = acc1b[i] + cb[16 + r];
    }
}

// ---------------------------------------------------------------------------
extern "C" void kernel_launch(void* const* d_in, const int* in_sizes, int n_in,
                              void* d_out, int out_size, void* d_ws, size_t ws_size,
                              hipStream_t stream)
{
    const float* x  = (const float*)d_in[0];
    // d_in[1], d_in[2] (fflow/bflow) unused: wt=0 -> same-frame search only
    const float* wq = (const float*)d_in[3];
    const float* bq = (const float*)d_in[4];
    const float* wk = (const float*)d_in[5];
    const float* bk = (const float*)d_in[6];
    const float* cw = (const float*)d_in[7];
    const float* cb = (const float*)d_in[8];
    float* out = (float*)d_out;

    char* ws = (char*)d_ws;
    _Float16*      xh   = (_Float16*)(ws + WS_XH);
    _Float16*      qbuf = (_Float16*)(ws + WS_Q);
    _Float16*      kbuf = (_Float16*)(ws + WS_K);
    float*         topw = (float*)   (ws + WS_TOPW);
    unsigned char* topi = (unsigned char*)(ws + WS_TOPI);
    _Float16*      awf  = (_Float16*)(ws + WS_AW);
    _Float16*      aqf  = (_Float16*)(ws + WS_AQ);
    _Float16*      akf  = (_Float16*)(ws + WS_AK);

    xcast_kernel  <<<NPIX / 256, 256, 0, stream>>>(x, xh);
    pack_kernel   <<<23, 256, 0, stream>>>(wq, wk, cw, awf, aqf, akf);
    // 262144 pixels -> 16384 16-pixel tiles; 4 waves (128 thr) per block
    proj_qk_kernel<<<4096, 128, 0, stream>>>(xh, aqf, akf, bq, bk, qbuf, kbuf);
    nls_topk_kernel<<<NPIX / 256, 256, 0, stream>>>(qbuf, kbuf, topw, topi);
    // conv: 32-pixel tiles, 4 waves/block -> 262144/128 = 2048 blocks
    nls_conv_kernel<<<2048, 128, 0, stream>>>(xh, awf, cb, topw, topi, out);
}
